// SurfaceConvPurity_15831249453210
// MI455X (gfx1250) — matrix operand order repaired and checked by reading, not compiled, not run on hardware
//
#include <hip/hip_runtime.h>

typedef __attribute__((ext_vector_type(16))) _Float16 v16h;
typedef __attribute__((ext_vector_type(8)))  _Float16 v8h;
typedef __attribute__((ext_vector_type(8)))  float    v8f;
typedef _Float16 half_t;

#define Bq   16
#define Nn   4096
#define NP   1024
#define Kk   32
#define ICH  64
#define OCH  64
#define CH   (ICH * OCH)          // 4096
#define NPTS (Bq * NP)            // 16384
#define S_WN (Bq * NP * Kk)       // 524288
#define EPSC 1e-5f

// ---------------------------------------------------------------- utilities

__global__ void k_zero(float* acc, int n) {
  int i = blockIdx.x * 256 + threadIdx.x;
  if (i < n) acc[i] = 0.0f;
}

__global__ void k_newxyz(const float* __restrict__ xyz, const int* __restrict__ didx,
                         float* __restrict__ out) {
  int i = blockIdx.x * 256 + threadIdx.x;           // NPTS*3
  if (i >= NPTS * 3) return;
  int s = i / 3, d = i % 3;
  int b = s / NP;
  int idx = didx[s];
  out[i] = xyz[(size_t)(b * Nn + idx) * 3 + d];
}

__global__ void k_prep_pts(const float* __restrict__ points, const float* __restrict__ xyz,
                           half_t* __restrict__ pts16) {
  int i = blockIdx.x * 256 + threadIdx.x;           // Bq*Nn*64
  if (i >= Bq * Nn * 64) return;
  int c = i & 63;
  int g = i >> 6;                                   // b*Nn + n
  float v = (c < 61) ? points[(size_t)g * 61 + c] : xyz[(size_t)g * 3 + (c - 61)];
  pts16[i] = (half_t)v;
}

__global__ void k_prep_lw(const float* __restrict__ lw, half_t* __restrict__ lwT) {
  int i = blockIdx.x * 256 + threadIdx.x;           // 64*CH
  if (i >= 64 * CH) return;
  int o = i / CH, c = i % CH;
  lwT[i] = (half_t)lw[(size_t)c * 64 + o];          // lw: [CH, 64] row-major
}

// ------------------------------------------------------- WeightNet statistics

__global__ void k_wn_stats0(const float* __restrict__ lc, const float* __restrict__ w0,
                            const float* __restrict__ b0,
                            float* __restrict__ gs, float* __restrict__ gq) {
  __shared__ float ls[16], lq[16];
  int t = threadIdx.x;
  if (t < 16) { ls[t] = 0.f; lq[t] = 0.f; }
  __syncthreads();
  int s = blockIdx.x * 256 + t;
  float x0 = lc[s * 3 + 0], x1 = lc[s * 3 + 1], x2 = lc[s * 3 + 2];
#pragma unroll
  for (int c = 0; c < 16; ++c) {
    float z = b0[c] + w0[c * 3] * x0 + w0[c * 3 + 1] * x1 + w0[c * 3 + 2] * x2;
    atomicAdd(&ls[c], z);
    atomicAdd(&lq[c], z * z);
  }
  __syncthreads();
  if (t < 16) { atomicAdd(&gs[t], ls[t]); atomicAdd(&gq[t], lq[t]); }
}

__global__ void k_wn_stats1(const float* __restrict__ lc,
                            const float* __restrict__ w0, const float* __restrict__ b0,
                            const float* __restrict__ g0, const float* __restrict__ be0,
                            const float* __restrict__ gs0, const float* __restrict__ gq0,
                            const float* __restrict__ w1, const float* __restrict__ b1,
                            float* __restrict__ gs1, float* __restrict__ gq1) {
  __shared__ float A0[16], B0[16], ls[16], lq[16], w1s[256];
  int t = threadIdx.x;
  if (t < 16) {
    float m = gs0[t] * (1.f / (float)S_WN);
    float v = gq0[t] * (1.f / (float)S_WN) - m * m;
    float inv = rsqrtf(v + EPSC);
    A0[t] = g0[t] * inv;
    B0[t] = be0[t] - g0[t] * m * inv;
    ls[t] = 0.f; lq[t] = 0.f;
  }
  w1s[t] = w1[t];
  __syncthreads();
  int s = blockIdx.x * 256 + t;
  float x0 = lc[s * 3], x1 = lc[s * 3 + 1], x2 = lc[s * 3 + 2];
  float h0[16];
#pragma unroll
  for (int c = 0; c < 16; ++c) {
    float z = b0[c] + w0[c * 3] * x0 + w0[c * 3 + 1] * x1 + w0[c * 3 + 2] * x2;
    float v = A0[c] * z + B0[c];
    h0[c] = v > 0.f ? v : 0.f;
  }
#pragma unroll
  for (int c = 0; c < 16; ++c) {
    float z = b1[c];
#pragma unroll
    for (int j = 0; j < 16; ++j) z += w1s[c * 16 + j] * h0[j];
    atomicAdd(&ls[c], z);
    atomicAdd(&lq[c], z * z);
  }
  __syncthreads();
  if (t < 16) { atomicAdd(&gs1[t], ls[t]); atomicAdd(&gq1[t], lq[t]); }
}

__global__ void k_wn_stats2(const float* __restrict__ lc,
                            const float* __restrict__ w0, const float* __restrict__ b0,
                            const float* __restrict__ g0, const float* __restrict__ be0,
                            const float* __restrict__ gs0, const float* __restrict__ gq0,
                            const float* __restrict__ w1, const float* __restrict__ b1,
                            const float* __restrict__ g1, const float* __restrict__ be1,
                            const float* __restrict__ gs1, const float* __restrict__ gq1,
                            const float* __restrict__ w2, const float* __restrict__ b2,
                            half_t* __restrict__ h1out,
                            float* __restrict__ gs2, float* __restrict__ gq2) {
  __shared__ float A0[16], B0[16], A1[16], B1[16];
  __shared__ float w1s[256], w2s[1024], ls[64], lq[64];
  int t = threadIdx.x;
  if (t < 16) {
    float m = gs0[t] * (1.f / (float)S_WN);
    float v = gq0[t] * (1.f / (float)S_WN) - m * m;
    float inv = rsqrtf(v + EPSC);
    A0[t] = g0[t] * inv;
    B0[t] = be0[t] - g0[t] * m * inv;
  } else if (t < 32) {
    int c = t - 16;
    float m = gs1[c] * (1.f / (float)S_WN);
    float v = gq1[c] * (1.f / (float)S_WN) - m * m;
    float inv = rsqrtf(v + EPSC);
    A1[c] = g1[c] * inv;
    B1[c] = be1[c] - g1[c] * m * inv;
  }
  if (t < 64) { ls[t] = 0.f; lq[t] = 0.f; }
  w1s[t] = w1[t];
  for (int i = t; i < 1024; i += 256) w2s[i] = w2[i];
  __syncthreads();
  int s = blockIdx.x * 256 + t;
  float x0 = lc[s * 3], x1 = lc[s * 3 + 1], x2 = lc[s * 3 + 2];
  float h0[16], h1v[16];
#pragma unroll
  for (int c = 0; c < 16; ++c) {
    float z = b0[c] + w0[c * 3] * x0 + w0[c * 3 + 1] * x1 + w0[c * 3 + 2] * x2;
    float v = A0[c] * z + B0[c];
    h0[c] = v > 0.f ? v : 0.f;
  }
#pragma unroll
  for (int c = 0; c < 16; ++c) {
    float z = b1[c];
#pragma unroll
    for (int j = 0; j < 16; ++j) z += w1s[c * 16 + j] * h0[j];
    float v = A1[c] * z + B1[c];
    h1v[c] = v > 0.f ? v : 0.f;
    h1out[(size_t)s * 16 + c] = (half_t)h1v[c];
  }
#pragma unroll
  for (int o = 0; o < 64; ++o) {
    float z = b2[o];
#pragma unroll
    for (int j = 0; j < 16; ++j) z += w2s[o * 16 + j] * h1v[j];
    atomicAdd(&ls[o], z);
    atomicAdd(&lq[o], z * z);
  }
  __syncthreads();
  if (t < 64) { atomicAdd(&gs2[t], ls[t]); atomicAdd(&gq2[t], lq[t]); }
}

// -------------------------------------------- per-point einsum via WMMA (f16)

#define PPB 16

__global__ __launch_bounds__(128)
void k_pointconv(const half_t* __restrict__ pts16, const int* __restrict__ nbr,
                 const half_t* __restrict__ h1,
                 const float* __restrict__ w2, const float* __restrict__ b2,
                 const float* __restrict__ g2, const float* __restrict__ be2,
                 const float* __restrict__ gs2, const float* __restrict__ gq2,
                 half_t* __restrict__ np1,
                 float* __restrict__ gs_np, float* __restrict__ gq_np) {
  __shared__ __align__(16) half_t Gt[64 * 32];   // Gt[i][k] = pts[nbr[k]][i]
  __shared__ __align__(16) half_t Wt[64 * 32];   // Wt[o][k] = weights[k][o]
  __shared__ __align__(16) half_t Dt[64 * 64];   // D tile staged for coalesced store
  __shared__ __align__(16) half_t w2t[64 * 32];  // w2 zero-padded to K=32, [o][j]
  __shared__ float A2[64], B2[64], b2s[64];

  const int t = threadIdx.x;
  const int lane = t & 31;
  const int wv = t >> 5;
  const int l15 = lane & 15;
  const int hi = (lane >= 16) ? 1 : 0;

  for (int i = t; i < 64 * 32; i += 128) {
    int o = i >> 5, j = i & 31;
    w2t[i] = (j < 16) ? (half_t)w2[o * 16 + j] : (half_t)0.f;
  }
  if (t < 64) {
    float m = gs2[t] * (1.f / (float)S_WN);
    float v = gq2[t] * (1.f / (float)S_WN) - m * m;
    float inv = rsqrtf(v + EPSC);
    A2[t] = g2[t] * inv;
    B2[t] = be2[t] - g2[t] * m * inv;
    b2s[t] = b2[t];
  }
  __syncthreads();

  float sum[4][8], ssq[4][8];
#pragma unroll
  for (int nt = 0; nt < 4; ++nt)
#pragma unroll
    for (int r = 0; r < 8; ++r) { sum[nt][r] = 0.f; ssq[nt][r] = 0.f; }

  const int p0 = blockIdx.x * PPB;
  for (int pt = 0; pt < PPB; ++pt) {
    const int p = p0 + pt;
    const int b = p / NP, n = p % NP;
    (void)n;
    if (pt + 1 < PPB) {
      __builtin_prefetch(nbr + (size_t)(p + 1) * Kk, 0, 0);
      __builtin_prefetch(h1 + (size_t)(p + 1) * Kk * 16, 0, 0);
    }
    {
      // stage Gt (gather + transpose): thread -> (k, 16-channel segment)
      int k = t & 31, seg = t >> 5;
      int idx = nbr[(size_t)p * Kk + k];
      const half_t* src = pts16 + ((size_t)(b * Nn + idx)) * 64 + seg * 16;
#pragma unroll
      for (int j = 0; j < 16; ++j) Gt[(seg * 16 + j) * 32 + k] = src[j];
    }
    {
      // stage Wt via WMMA: z2[k][o] = h1[p,k,:] . w2[o,:]  (K=16 padded to 32)
      // M = k rows (2 tiles), N = o (4 tiles); wave -> (mt = wv&1, ntile pair wv>>1)
      const int mt = wv & 1;
      const int ng = wv >> 1;
      const int krow = mt * 16 + l15;
      const half_t* hp = h1 + ((size_t)p * Kk + krow) * 16 + hi * 8;
      v16h a = {};                          // e>=8 stays 0 (K padding)
#pragma unroll
      for (int e = 0; e < 8; ++e) a[e] = hp[e];
      const int kbB = hi * 8;               // table order; e>=8 reads zero pad
      const int kbase = mt * 16 + hi * 8;   // 8 contiguous k per lane in D layout
#pragma unroll
      for (int nt2 = 0; nt2 < 2; ++nt2) {
        int o = (ng * 2 + nt2) * 16 + l15;
        v16h bvec;
#pragma unroll
        for (int e = 0; e < 16; ++e) bvec[e] = w2t[o * 32 + kbB + e + (e >= 8 ? 8 : 0)];
        v8f d = {};
        d = __builtin_amdgcn_wmma_f32_16x16x32_f16(false, a, false, bvec,
                                                   (short)0, d, false, false);
        v8h w8;
#pragma unroll
        for (int r = 0; r < 8; ++r) {
          float z = d[r] + b2s[o];
          float v = A2[o] * z + B2[o];
          w8[r] = (half_t)(v > 0.f ? v : 0.f);
        }
        *(v8h*)&Wt[o * 32 + kbase] = w8;    // one ds_store_b128 per lane
      }
    }
    __syncthreads();
    {
      // main WMMA: out[i][o] = sum_k Gt[i][k] * Wt[o][k]
      const int m = wv * 16 + l15;
      const int kbA = hi * 8;
      v16h a;
#pragma unroll
      for (int e = 0; e < 8; ++e) {
        a[e]     = Gt[m * 32 + kbA + e];
        a[8 + e] = Gt[m * 32 + kbA + 16 + e];
      }
      const int kbB = hi * 8;
#pragma unroll
      for (int nt = 0; nt < 4; ++nt) {
        int nn = nt * 16 + l15;
        v16h bb;
#pragma unroll
        for (int e = 0; e < 16; ++e) bb[e] = Wt[nn * 32 + kbB + e + (e >= 8 ? 8 : 0)];
        v8f c = {};
        c = __builtin_amdgcn_wmma_f32_16x16x32_f16(false, a, false, bb,
                                                   (short)0, c, false, false);
#pragma unroll
        for (int r = 0; r < 8; ++r) {
          float v = c[r];
          int row = wv * 16 + r + hi * 8;   // IN_CH index i
          Dt[row * 64 + nn] = (half_t)v;
          sum[nt][r] += v;
          ssq[nt][r] += v * v;
        }
      }
    }
    __syncthreads();
    {
      // coalesced copy Dt -> np1 (32 contiguous halves per thread, b128 stores)
      half_t* dst = np1 + (size_t)p * CH + t * 32;
      const half_t* srcl = &Dt[t * 32];
#pragma unroll
      for (int j = 0; j < 4; ++j)
        *(v8h*)(dst + j * 8) = *(const v8h*)(srcl + j * 8);
    }
    // no barrier needed: next compute (which rewrites Dt) is fenced by the
    // staging barrier above; staging only touches Gt/Wt.
  }
#pragma unroll
  for (int nt = 0; nt < 4; ++nt)
#pragma unroll
    for (int r = 0; r < 8; ++r) {
      int row = wv * 16 + r + hi * 8;
      int col = nt * 16 + l15;
      int ch = row * 64 + col;
      atomicAdd(&gs_np[ch], sum[nt][r]);
      atomicAdd(&gq_np[ch], ssq[nt][r]);
    }
}

// ---------------------------------------------------- BN coefficient kernel

__global__ void k_bn_coef(const float* __restrict__ gs, const float* __restrict__ gq,
                          const float* __restrict__ gam, const float* __restrict__ bet,
                          float* __restrict__ A, float* __restrict__ Bc,
                          int nch, float invcnt) {
  int c = blockIdx.x * 256 + threadIdx.x;
  if (c >= nch) return;
  float m = gs[c] * invcnt;
  float v = gq[c] * invcnt - m * m;
  float inv = rsqrtf(v + EPSC);
  A[c] = gam[c] * inv;
  Bc[c] = bet[c] - gam[c] * m * inv;
}

// ------------------------------------------- big linear [16384,4096]x[4096,64]

__global__ __launch_bounds__(128)
void k_linear(const half_t* __restrict__ np1,
              const float* __restrict__ Anp, const float* __restrict__ Bnp,
              const half_t* __restrict__ lwT, const float* __restrict__ lb,
              float* __restrict__ y,
              float* __restrict__ gs_y, float* __restrict__ gq_y) {
  __shared__ __align__(16) half_t Al[64 * 32];
  __shared__ float ca[32], cb[32];
  const int t = threadIdx.x;
  const int lane = t & 31, wv = t >> 5;
  const int l15 = lane & 15;
  const int hi = (lane >= 16) ? 1 : 0;
  const int s0 = blockIdx.x * 64;

  v8f acc[4];
#pragma unroll
  for (int nt = 0; nt < 4; ++nt) { v8f z = {}; acc[nt] = z; }

  for (int kc = 0; kc < CH; kc += 32) {
    if (t < 32) { ca[t] = Anp[kc + t]; cb[t] = Bnp[kc + t]; }
    __syncthreads();
    {
      int row = t >> 1, hh = t & 1;
      const half_t* src = np1 + (size_t)(s0 + row) * CH + kc + hh * 16;
#pragma unroll
      for (int j = 0; j < 16; ++j) {
        int c = hh * 16 + j;
        float x = (float)src[j];
        float v = ca[c] * x + cb[c];
        v = v > 0.f ? v : 0.f;                // BN + relu fused on load
        Al[row * 32 + c] = (half_t)v;
      }
    }
    __syncthreads();
    const int m = wv * 16 + l15;
    const int kbA = hi * 8;
    v16h a;
#pragma unroll
    for (int e = 0; e < 8; ++e) {
      a[e]     = Al[m * 32 + kbA + e];
      a[8 + e] = Al[m * 32 + kbA + 16 + e];
    }
    const int kbB = hi * 8;
#pragma unroll
    for (int nt = 0; nt < 4; ++nt) {
      int o = nt * 16 + l15;
      const half_t* bp = lwT + (size_t)o * CH + kc + kbB;
      v16h bb;
#pragma unroll
      for (int e = 0; e < 16; ++e) bb[e] = bp[e + (e >= 8 ? 8 : 0)];
      acc[nt] = __builtin_amdgcn_wmma_f32_16x16x32_f16(false, a, false, bb,
                                                       (short)0, acc[nt], false, false);
    }
    __syncthreads();
  }
#pragma unroll
  for (int nt = 0; nt < 4; ++nt) {
    int o = nt * 16 + l15;
    float bias = lb[o];
    float s = 0.f, sq = 0.f;
#pragma unroll
    for (int r = 0; r < 8; ++r) {
      float v = acc[nt][r] + bias;
      int row = s0 + wv * 16 + r + hi * 8;
      y[(size_t)row * 64 + o] = v;
      s += v;
      sq += v * v;
    }
    atomicAdd(&gs_y[o], s);
    atomicAdd(&gq_y[o], sq);
  }
}

__global__ void k_out(const float* __restrict__ y,
                      const float* __restrict__ gs_y, const float* __restrict__ gq_y,
                      const float* __restrict__ gl, const float* __restrict__ bl,
                      float* __restrict__ out) {
  int i = blockIdx.x * 256 + threadIdx.x;     // NPTS*64
  if (i >= NPTS * 64) return;
  int o = i & 63;
  float m = gs_y[o] * (1.f / (float)NPTS);
  float v = gq_y[o] * (1.f / (float)NPTS) - m * m;
  float inv = rsqrtf(v + EPSC);
  float val = gl[o] * (y[i] - m) * inv + bl[o];
  out[i] = val > 0.f ? val : 0.f;
}

// -------------------------------------------------------------------- launch

extern "C" void kernel_launch(void* const* d_in, const int* in_sizes, int n_in,
                              void* d_out, int out_size, void* d_ws, size_t ws_size,
                              hipStream_t stream) {
  const float* xyz    = (const float*)d_in[0];
  const float* points = (const float*)d_in[1];
  const float* lc     = (const float*)d_in[2];
  const int*   nbr    = (const int*)d_in[3];
  const int*   didx   = (const int*)d_in[4];
  const float* w0  = (const float*)d_in[5];
  const float* b0  = (const float*)d_in[6];
  const float* g0  = (const float*)d_in[7];
  const float* be0 = (const float*)d_in[8];
  const float* w1  = (const float*)d_in[9];
  const float* b1  = (const float*)d_in[10];
  const float* g1  = (const float*)d_in[11];
  const float* be1 = (const float*)d_in[12];
  const float* w2  = (const float*)d_in[13];
  const float* b2  = (const float*)d_in[14];
  const float* g2  = (const float*)d_in[15];
  const float* be2 = (const float*)d_in[16];
  const float* gc  = (const float*)d_in[17];
  const float* bc  = (const float*)d_in[18];
  const float* lw  = (const float*)d_in[19];
  const float* lb  = (const float*)d_in[20];
  const float* gl  = (const float*)d_in[21];
  const float* bl  = (const float*)d_in[22];
  float* out = (float*)d_out;

  char* ws = (char*)d_ws;
  size_t off = 0;
  auto walloc = [&](size_t bytes) -> char* {
    char* p = ws + off;
    off = (off + bytes + 255) & ~(size_t)255;
    return p;
  };
  half_t* pts16 = (half_t*)walloc((size_t)Bq * Nn * 64 * 2);
  half_t* h1    = (half_t*)walloc((size_t)S_WN * 16 * 2);
  half_t* np1   = (half_t*)walloc((size_t)NPTS * CH * 2);
  half_t* lwT   = (half_t*)walloc((size_t)64 * CH * 2);
  float*  yb    = (float*)walloc((size_t)NPTS * 64 * 4);
  const int NACC = 16 * 4 + 64 * 2 + 4096 * 2 + 64 * 2;   // 8512
  float* acc = (float*)walloc((size_t)NACC * 4);
  float* gs0 = acc,        * gq0 = gs0 + 16;
  float* gs1 = gq0 + 16,   * gq1 = gs1 + 16;
  float* gs2 = gq1 + 16,   * gq2 = gs2 + 64;
  float* gs_np = gq2 + 64, * gq_np = gs_np + 4096;
  float* gs_y = gq_np + 4096, * gq_y = gs_y + 64;
  float* Anp = (float*)walloc((size_t)CH * 4);
  float* Bnp = (float*)walloc((size_t)CH * 4);

  k_zero<<<(NACC + 255) / 256, 256, 0, stream>>>(acc, NACC);
  k_newxyz<<<(NPTS * 3 + 255) / 256, 256, 0, stream>>>(xyz, didx, out);
  k_prep_pts<<<(Bq * Nn * 64) / 256, 256, 0, stream>>>(points, xyz, pts16);
  k_prep_lw<<<(64 * CH) / 256, 256, 0, stream>>>(lw, lwT);

  k_wn_stats0<<<S_WN / 256, 256, 0, stream>>>(lc, w0, b0, gs0, gq0);
  k_wn_stats1<<<S_WN / 256, 256, 0, stream>>>(lc, w0, b0, g0, be0, gs0, gq0,
                                              w1, b1, gs1, gq1);
  k_wn_stats2<<<S_WN / 256, 256, 0, stream>>>(lc, w0, b0, g0, be0, gs0, gq0,
                                              w1, b1, g1, be1, gs1, gq1,
                                              w2, b2, h1, gs2, gq2);

  k_pointconv<<<NPTS / PPB, 128, 0, stream>>>(pts16, nbr, h1, w2, b2, g2, be2,
                                              gs2, gq2, np1, gs_np, gq_np);

  k_bn_coef<<<(CH + 255) / 256, 256, 0, stream>>>(gs_np, gq_np, gc, bc, Anp, Bnp,
                                                  CH, 1.f / (float)NPTS);

  k_linear<<<NPTS / 64, 128, 0, stream>>>(np1, Anp, Bnp, lwT, lb, yb, gs_y, gq_y);

  k_out<<<(NPTS * 64) / 256, 256, 0, stream>>>(yb, gs_y, gq_y, gl, bl, out + NPTS * 3);
}
